// LocalConnect_78847009620385
// MI455X (gfx1250) — compile-verified, matrix operand order checked by reading, not run on hardware
//
#include <hip/hip_runtime.h>

typedef __attribute__((ext_vector_type(16))) __bf16        v16bf;
typedef __attribute__((ext_vector_type(8)))  float         v8f;
typedef __attribute__((ext_vector_type(4)))  unsigned int  u32x4;
typedef __attribute__((ext_vector_type(8)))  int           i32x8;
typedef __attribute__((ext_vector_type(4)))  int           i32x4;

union Frag {
    u32x4 q[2];   // two 16-byte chunks (8 bf16 each)
    v16bf v;      // 32-byte WMMA operand
};

__device__ __forceinline__ unsigned short f2bf(float f) {
    unsigned u = __builtin_bit_cast(unsigned, f);
    unsigned r = (u + 0x7FFFu + ((u >> 16) & 1u)) >> 16;   // RNE f32->bf16
    return (unsigned short)r;
}

#define CH_IN   64
#define CH_OUT  128
#define IMG_H   112
#define IMG_W   112
#define PADW    114          // -1 .. 112 padded columns
#define CSTRIDE 72           // padded channel stride (halves): 32 DW row + 4 DW pad

#if defined(__HIP_DEVICE_COMPILE__) && __has_builtin(__builtin_amdgcn_tensor_load_to_lds)
#define HAVE_TDM 1
#else
#define HAVE_TDM 0
#endif

// ---- one-shot weight precondition: f32 (ko,ci,kh,kw) -> bf16 tap-major (r,ko,ci) ----
__global__ __launch_bounds__(256)
void prep_weights(const float* __restrict__ wgt, unsigned short* __restrict__ wbf)
{
    int i = blockIdx.x * 256 + threadIdx.x;           // i = (ko*64+ci)*9 + r
    if (i < CH_OUT * CH_IN * 9) {
        int r    = i % 9;
        int koci = i / 9;
        wbf[r * (CH_OUT * CH_IN) + koci] = f2bf(wgt[i]);
    }
}

// ---- TDM: DMA one 128x64 bf16 weight tap into LDS with 4-DW pad per 32-DW row ----
__device__ __forceinline__ void tdm_copy_tap(const unsigned short* gsrc, unsigned lds_off)
{
#if HAVE_TDM
    unsigned long long ga = (unsigned long long)(size_t)gsrc;
    u32x4 g0;
    g0.x = 1u;                                        // count=1, load, no gather
    g0.y = lds_off;                                   // LDS byte address
    g0.z = (unsigned)(ga & 0xFFFFFFFFu);              // global_addr[31:0]
    g0.w = (unsigned)((ga >> 32) & 0x01FFFFFFu)       // global_addr[56:32]
         | (2u << 30);                                // type = 2 ("image")
    i32x8 g1;
    g1[0] = (int)((1u << 16)                          // data_size = 2 bytes
                | (1u << 20)                          // pad_enable
                | (4u << 22)                          // pad_interval: 32 DWORDs
                | (3u << 25));                        // pad_amount : 4 DWORDs
    g1[1] = (int)(64u  << 16);                        // tensor_dim0 = 64  [63:48]
    g1[2] = (int)(128u << 16);                        // tensor_dim1 = 128 [95:80]
    g1[3] = (int)(64u  << 16);                        // tile_dim0   = 64  [127:112]
    g1[4] = 128;                                      // tile_dim1   = 128 [143:128]
    g1[5] = 64;                                       // tensor_dim0_stride = 64
    g1[6] = 0;
    g1[7] = 0;
    i32x4 z4 = {0, 0, 0, 0};
#if __clang_major__ >= 23
    i32x8 z8 = {0, 0, 0, 0, 0, 0, 0, 0};
    __builtin_amdgcn_tensor_load_to_lds(g0, g1, z4, z4, z8, 0);
#else
    __builtin_amdgcn_tensor_load_to_lds(g0, g1, z4, z4, 0);
#endif
#endif
}

// MODE 0: weights from f32 global (no workspace). MODE 1: bf16 tap-major from d_ws (+TDM).
template<int MODE>
__global__ __launch_bounds__(256)
void conv3x3_wmma_bf16(const float* __restrict__ x,
                       const float* __restrict__ wgt,
                       const unsigned short* __restrict__ wbf,
                       float* __restrict__ out)
{
    __shared__ __align__(16) unsigned short xs[3 * PADW * CSTRIDE];      // ~49 KB
    __shared__ __align__(16) unsigned short wts[2][CH_OUT * CSTRIDE];    // 2 x 18 KB

    const int h    = blockIdx.x % IMG_H;
    const int nimg = blockIdx.x / IMG_H;
    const int tid  = threadIdx.x;
    const int wave = tid >> 5;
    const int lane = tid & 31;
    const int l    = lane & 15;      // row(A)/col(B) within 16x16 tile
    const int hi   = lane >> 4;      // K sub-chunk select (+8)
    const int m0   = wave * 16;

    // ---- stage input rows h-1..h+1 into LDS as bf16 (zero-padded) ----
    const float* xbase = x + (size_t)nimg * CH_IN * IMG_H * IMG_W;
    for (int idx = tid; idx < CH_IN * 3 * PADW; idx += 256) {
        const int ci  = idx / (3 * PADW);
        const int rem = idx % (3 * PADW);
        const int row = rem / PADW;
        const int col = rem % PADW;
        const int grow = h + row - 1;
        const int gcol = col - 1;
        float v = 0.0f;
        if ((unsigned)grow < (unsigned)IMG_H && (unsigned)gcol < (unsigned)IMG_W)
            v = xbase[((size_t)ci * IMG_H + grow) * IMG_W + gcol];
        xs[(row * PADW + col) * CSTRIDE + ci] = f2bf(v);
    }

    // ---- weight-tap staging helpers ----
    auto stage_start = [&](int r, int nb) {
        if (MODE == 1 && HAVE_TDM) {
            if (wave == 0)
                tdm_copy_tap(wbf + r * (CH_OUT * CH_IN),
                             (unsigned)(size_t)&wts[nb][0]);
        } else if (MODE == 1) {
            const u32x4* src = (const u32x4*)(wbf + r * (CH_OUT * CH_IN));
            for (int i = tid; i < CH_OUT * CH_IN / 8; i += 256) {    // 1024 x 16B
                const int ko = i >> 3, c8 = i & 7;
                *(u32x4*)&wts[nb][ko * CSTRIDE + c8 * 8] = src[i];
            }
        } else {
            for (int idx = tid; idx < CH_OUT * CH_IN; idx += 256)
                wts[nb][(idx >> 6) * CSTRIDE + (idx & 63)] = f2bf(wgt[idx * 9 + r]);
            if (r < 8)
                __builtin_prefetch(&wgt[(tid * 32) * 9 + r + 1], 0, 1);
        }
    };
    auto stage_finish = [&]() {
#if HAVE_TDM
        if (MODE == 1 && wave == 0)
            __builtin_amdgcn_s_wait_tensorcnt(0);
#endif
    };

    v8f acc[7];
#pragma unroll
    for (int t = 0; t < 7; ++t) acc[t] = 0.0f;

    // prologue: tap 0 into buffer 0
    stage_start(0, 0);
    stage_finish();
    __syncthreads();

    for (int r = 0; r < 9; ++r) {
        const int cur = r & 1;
        if (r < 8) stage_start(r + 1, cur ^ 1);      // DMA next tap during compute

        const int kh = r / 3;
        const int kw = r % 3;
        const unsigned short* wcur = &wts[cur][0];
        const unsigned short* xrow = &xs[kh * PADW * CSTRIDE];
        const int ko8 = hi * 8;

        // A fragments for both K-chunks up front
        Frag a0, a1;
        {
            const u32x4* wp = (const u32x4*)&wcur[(m0 + l) * CSTRIDE + ko8];
            a0.q[0] = wp[0];
            a0.q[1] = wp[2];
            a1.q[0] = wp[4];          // +32 halves (c0 = 32)
            a1.q[1] = wp[6];
        }

        // B fragments, chunk 0, batched
        Frag b[7];
#pragma unroll
        for (int t = 0; t < 7; ++t) {
            const u32x4* xp =
                (const u32x4*)&xrow[(t * 16 + l + kw) * CSTRIDE + ko8];
            b[t].q[0] = xp[0];
            b[t].q[1] = xp[2];
        }
        // chunk-0 WMMA; refill b[t] with chunk-1 data right after consumption so
        // each chunk-1 ds_load has ~6 WMMAs of slack before its use
#pragma unroll
        for (int t = 0; t < 7; ++t) {
            acc[t] = __builtin_amdgcn_wmma_f32_16x16x32_bf16(
                false, a0.v, false, b[t].v, (short)0, acc[t], false, false);
            const u32x4* xp =
                (const u32x4*)&xrow[(t * 16 + l + kw) * CSTRIDE + 32 + ko8];
            b[t].q[0] = xp[0];
            b[t].q[1] = xp[2];
        }
#pragma unroll
        for (int t = 0; t < 7; ++t)
            acc[t] = __builtin_amdgcn_wmma_f32_16x16x32_bf16(
                false, a1.v, false, b[t].v, (short)0, acc[t], false, false);

        if (r < 8) {
            stage_finish();          // issuing wave drains TENSORcnt
            __syncthreads();         // publish next tap to all waves
        }
    }

    // ---- store: elem e of lane -> m = m0 + hi*8 + e, n = t*16 + l (== w) ----
    float* obase = out + (size_t)nimg * CH_OUT * IMG_H * IMG_W + (size_t)h * IMG_W;
#pragma unroll
    for (int t = 0; t < 7; ++t) {
        const int n = t * 16 + l;
#pragma unroll
        for (int e = 0; e < 8; ++e) {
            const int m = m0 + hi * 8 + e;
            obase[(size_t)m * IMG_H * IMG_W + n] = acc[t][e];
        }
    }
}

extern "C" void kernel_launch(void* const* d_in, const int* in_sizes, int n_in,
                              void* d_out, int out_size, void* d_ws, size_t ws_size,
                              hipStream_t stream)
{
    const float* x = (const float*)d_in[0];   // (32,64,112,112)
    const float* w = (const float*)d_in[1];   // (128,64,3,3)
    float* out = (float*)d_out;               // (32,128,112,112)

    const size_t WBF_BYTES = (size_t)9 * CH_OUT * CH_IN * sizeof(unsigned short);
    dim3 grid(32 * IMG_H);

    if (ws_size >= WBF_BYTES) {
        unsigned short* wbf = (unsigned short*)d_ws;
        const int n = 9 * CH_OUT * CH_IN;
        prep_weights<<<(n + 255) / 256, 256, 0, stream>>>(w, wbf);
        conv3x3_wmma_bf16<1><<<grid, 256, 0, stream>>>(x, w, wbf, out);
    } else {
        conv3x3_wmma_bf16<0><<<grid, 256, 0, stream>>>(x, w, nullptr, out);
    }
}
